// Seq2Seq_80616536146603
// MI455X (gfx1250) — compile-verified, matrix operand order checked by reading, not run on hardware
//
#include <hip/hip_runtime.h>
#include <hip/hip_bf16.h>

typedef __attribute__((ext_vector_type(8)))  float  v8f;
typedef __attribute__((ext_vector_type(16))) __bf16 v16bf;

// ---------------- problem constants ----------------
constexpr int Hd = 1024;       // hidden  (== K == lda == ldw for every GEMM)
constexpr int Bd = 16;         // batch
constexpr int Sd = 512;        // src len
constexpr int Td = 256;        // trg len
constexpr int Vd = 32000;      // vocab
constexpr int Gd = 4 * Hd;     // 4H gates
constexpr int HP = Hd + 8;     // LDS row pitch (bf16) to avoid bank conflicts

// ---------------- WMMA fragment load ----------------
// 16-bit A/B fragment layout (ISA 7.12.2): per lane, row r, two 8-elem
// (16-byte) K-chunks at kk + 8*(lane>>4) and kk + 16 + 8*(lane>>4).
// `off` is a compile-time-constant element offset that folds into the
// instruction's 24-bit immediate (sibling rows cost no address VGPRs).
__device__ __forceinline__ v16bf load_frag(const __bf16* rowPtr, int off,
                                           int kk, int lane) {
  int kb = (lane >> 4) << 3;
  union { v16bf v; uint4 u[2]; } t;
  t.u[0] = *reinterpret_cast<const uint4*>(rowPtr + off + kk + kb);
  t.u[1] = *reinterpret_cast<const uint4*>(rowPtr + off + kk + 16 + kb);
  return t.v;
}

#define WMMA_BF16(A, B, C) \
  __builtin_amdgcn_wmma_f32_16x16x32_bf16(false, (A), false, (B), (short)0, (C), false, false)

__device__ __forceinline__ float fsigmoid(float x) { return 1.0f / (1.0f + __expf(-x)); }
__device__ __forceinline__ float ftanh(float x) {
  float e = __expf(2.0f * x);
  return (e - 1.0f) / (e + 1.0f);
}

// ---------------- small utility kernels ----------------
__global__ void k_f32_to_bf16(const float* __restrict__ in,
                              __bf16* __restrict__ out, long n) {
  long stride = (long)gridDim.x * blockDim.x;
  for (long i = blockIdx.x * (long)blockDim.x + threadIdx.x; i < n; i += stride)
    out[i] = (__bf16)in[i];
}

__global__ void k_zero_bf16(__bf16* __restrict__ p, long n) {
  long stride = (long)gridDim.x * blockDim.x;
  for (long i = blockIdx.x * (long)blockDim.x + threadIdx.x; i < n; i += stride)
    p[i] = (__bf16)0.0f;
}

__global__ void k_zero_f32(float* __restrict__ p, long n) {
  long stride = (long)gridDim.x * blockDim.x;
  for (long i = blockIdx.x * (long)blockDim.x + threadIdx.x; i < n; i += stride)
    p[i] = 0.0f;
}

__global__ void k_zero_t0(float* __restrict__ out) {
  long n = (long)Bd * Vd;
  long stride = (long)gridDim.x * blockDim.x;
  for (long i = blockIdx.x * (long)blockDim.x + threadIdx.x; i < n; i += stride) {
    int b = (int)(i / Vd), v = (int)(i % Vd);
    out[(size_t)b * Td * Vd + v] = 0.0f;
  }
}

// Embedding gather: out[(t*B + b)*H + h] = bf16(E[tok[b*tokLd + t]*H + h])
__global__ void k_embed(const int* __restrict__ tok, int tokLd,
                        const float* __restrict__ E,
                        __bf16* __restrict__ out, long total) {
  long stride = (long)gridDim.x * blockDim.x;
  for (long i = blockIdx.x * (long)blockDim.x + threadIdx.x; i < total; i += stride) {
    int hh = (int)(i % Hd);
    long r = i / Hd;
    int b = (int)(r % Bd);
    int t = (int)(r / Bd);
    int token = tok[(size_t)b * tokLd + t];
    out[i] = (__bf16)E[(size_t)token * Hd + hh];
  }
}

// ---------------- GEMM core: 32x64 tile, K = Hd, strides = Hd ---------------
// 2 A-frags x 4 B-frags from two base pointers (+ constant offsets), 8 accs.
__device__ __forceinline__ void gemm_tile_32x64(const __bf16* __restrict__ aPtr,
                                                const __bf16* __restrict__ wPtr,
                                                int lane, v8f acc[2][4]) {
  for (int kk = 0; kk < Hd; kk += 32) {
    v16bf a0 = load_frag(aPtr, 0,       kk, lane);
    v16bf a1 = load_frag(aPtr, 16 * Hd, kk, lane);
    v16bf b0 = load_frag(wPtr, 0,       kk, lane);
    v16bf b1 = load_frag(wPtr, 16 * Hd, kk, lane);
    v16bf b2 = load_frag(wPtr, 32 * Hd, kk, lane);
    v16bf b3 = load_frag(wPtr, 48 * Hd, kk, lane);
    acc[0][0] = WMMA_BF16(a0, b0, acc[0][0]);
    acc[0][1] = WMMA_BF16(a0, b1, acc[0][1]);
    acc[0][2] = WMMA_BF16(a0, b2, acc[0][2]);
    acc[0][3] = WMMA_BF16(a0, b3, acc[0][3]);
    acc[1][0] = WMMA_BF16(a1, b0, acc[1][0]);
    acc[1][1] = WMMA_BF16(a1, b1, acc[1][1]);
    acc[1][2] = WMMA_BF16(a1, b2, acc[1][2]);
    acc[1][3] = WMMA_BF16(a1, b3, acc[1][3]);
  }
}

// ---- gx GEMM: C(bf16, MxGd) = A(MxHd) * W(GdxHd)^T + bih + bhh -------------
__global__ void __launch_bounds__(256)
k_gemm32(const __bf16* __restrict__ A, const __bf16* __restrict__ W,
         const float* __restrict__ bias0, const float* __restrict__ bias1,
         __bf16* __restrict__ C, int M, int Mtiles32, int Ntiles64) {
  int wid  = (int)((blockIdx.x * (long)blockDim.x + threadIdx.x) >> 5);
  if (wid >= Mtiles32 * Ntiles64) return;
  int lane = threadIdx.x & 31;
  int mt = wid / Ntiles64;
  int nt = wid % Ntiles64;
  int m0 = mt * 32;
  if (m0 + 32 > M) m0 = M - 32;       // overlap last tile (same values rewritten)
  int n0 = nt * 64;
  int nl = lane & 15;

  const __bf16* aPtr = A + (size_t)(m0 + nl) * Hd;
  const __bf16* wPtr = W + (size_t)(n0 + nl) * Hd;

  v8f acc[2][4] = {};
  gemm_tile_32x64(aPtr, wPtr, lane, acc);

  int hi = lane >> 4;
#pragma unroll
  for (int mi = 0; mi < 2; ++mi) {
#pragma unroll
    for (int j = 0; j < 4; ++j) {
      int n = n0 + j * 16 + nl;
      float badd = 0.0f;
      if (bias0) badd += bias0[n];
      if (bias1) badd += bias1[n];
#pragma unroll
      for (int r = 0; r < 8; ++r) {
        int m = m0 + mi * 16 + r + 8 * hi;
        C[(size_t)m * Gd + n] = (__bf16)(acc[mi][j][r] + badd);
      }
    }
  }
}

// ---- FC GEMM with (B,T,V) scatter + bias, fp32 out -------------------------
// A = hs (rows m = t*16 + b), out[b, t+1, n] = sum_k A[m,k]*W[n,k] + bias[n]
__global__ void __launch_bounds__(256)
k_fc32(const __bf16* __restrict__ A, const __bf16* __restrict__ W,
       const float* __restrict__ bias, float* __restrict__ out,
       int M, int Mtiles32, int Ntiles64) {
  int wid  = (int)((blockIdx.x * (long)blockDim.x + threadIdx.x) >> 5);
  if (wid >= Mtiles32 * Ntiles64) return;
  int lane = threadIdx.x & 31;
  int mt = wid / Ntiles64;
  int nt = wid % Ntiles64;
  int m0 = mt * 32;
  if (m0 + 32 > M) m0 = M - 32;       // M = 4080: last tile overlaps, rows stay valid
  int n0 = nt * 64;
  int nl = lane & 15;

  const __bf16* aPtr = A + (size_t)(m0 + nl) * Hd;
  const __bf16* wPtr = W + (size_t)(n0 + nl) * Hd;

  v8f acc[2][4] = {};
  gemm_tile_32x64(aPtr, wPtr, lane, acc);

  int hi = lane >> 4;
#pragma unroll
  for (int mi = 0; mi < 2; ++mi) {
#pragma unroll
    for (int j = 0; j < 4; ++j) {
      int n = n0 + j * 16 + nl;
      float badd = bias[n];
#pragma unroll
      for (int r = 0; r < 8; ++r) {
        int m = m0 + mi * 16 + r + 8 * hi;
        int t = m >> 4;               // row m = t*16 + b
        int b = m & 15;
        out[(size_t)b * Td * Vd + (size_t)(t + 1) * Vd + n] = acc[mi][j][r] + badd;
      }
    }
  }
}

// ---------------- fused LSTM recurrent step (split-K x4, LDS) ---------------
// 32 blocks x 8 waves. Each block handles 2 hidden-column tiles (16 cols);
// wave w: ntile = w>>2, ksplit = w&3 covers K/4 of h @ Whh^T for all 4 gates.
__global__ void __launch_bounds__(256)
k_lstm_step(const __bf16* __restrict__ gx, const __bf16* __restrict__ hin,
            const __bf16* __restrict__ Whh, float* __restrict__ c,
            __bf16* __restrict__ hout, __bf16* __restrict__ hs) {
  __shared__ __bf16 hsh[Bd * HP];                 // staged h, padded rows
  __shared__ float partial[8][4][32][9];          // [wave][gate][lane][r(+pad)]
  __shared__ float gsum[2][4][32][9];             // reduced gates per ntile

  int tid  = threadIdx.x;
  int lane = tid & 31;
  int w    = tid >> 5;                            // 0..7
  int nl   = lane & 15;
  int hi   = lane >> 4;

  // cooperative copy of h (16 x 1024 bf16) into padded LDS rows, 16B chunks
  {
    const uint4* hg = reinterpret_cast<const uint4*>(hin);
    for (int i = tid; i < (Bd * Hd) / 8; i += 256) {
      int row = i >> 7;                           // 128 uint4 per row
      int col = i & 127;
      *(reinterpret_cast<uint4*>(&hsh[row * HP]) + col) = hg[i];
    }
  }
  __syncthreads();

  int ntile = blockIdx.x * 2 + (w >> 2);          // 0..63
  int ks    = w & 3;
  int n0    = ntile * 16;
  int k0    = ks * (Hd / 4);

  const __bf16* aRow = &hsh[nl * HP];
  const __bf16* wPtr = Whh + (size_t)(n0 + nl) * Hd;   // gate rows at +g*Hd*Hd

  v8f acc[4] = {};
  for (int kk = k0; kk < k0 + Hd / 4; kk += 32) {
    v16bf a  = load_frag(aRow, 0, kk, lane);      // ds_load_b128 from LDS
    v16bf bI = load_frag(wPtr, 0 * Hd * Hd, kk, lane);
    v16bf bF = load_frag(wPtr, 1 * Hd * Hd, kk, lane);
    v16bf bG = load_frag(wPtr, 2 * Hd * Hd, kk, lane);
    v16bf bO = load_frag(wPtr, 3 * Hd * Hd, kk, lane);
    acc[0] = WMMA_BF16(a, bI, acc[0]);
    acc[1] = WMMA_BF16(a, bF, acc[1]);
    acc[2] = WMMA_BF16(a, bG, acc[2]);
    acc[3] = WMMA_BF16(a, bO, acc[3]);
  }

#pragma unroll
  for (int g = 0; g < 4; ++g)
#pragma unroll
    for (int r = 0; r < 8; ++r)
      partial[w][g][lane][r] = acc[g][r];
  __syncthreads();

  // phase 1: 8 waves reduce the 4 K-splits; wave w -> (nt = w&1, gate = w>>1)
  {
    int nt = w & 1, gg = w >> 1;
#pragma unroll
    for (int r = 0; r < 8; ++r) {
      float s = partial[nt * 4 + 0][gg][lane][r] + partial[nt * 4 + 1][gg][lane][r]
              + partial[nt * 4 + 2][gg][lane][r] + partial[nt * 4 + 3][gg][lane][r];
      gsum[nt][gg][lane][r] = s;
    }
  }
  __syncthreads();

  // phase 2: waves 0..1 apply gate nonlinearity and update c/h
  if (w < 2) {
    int n = (blockIdx.x * 2 + w) * 16 + nl;
#pragma unroll
    for (int r = 0; r < 8; ++r) {
      int m = r + 8 * hi;                         // batch row
      size_t gb = (size_t)m * Gd;
      float iv = gsum[w][0][lane][r] + (float)gx[gb + 0 * Hd + n];
      float fv = gsum[w][1][lane][r] + (float)gx[gb + 1 * Hd + n];
      float gv = gsum[w][2][lane][r] + (float)gx[gb + 2 * Hd + n];
      float ov = gsum[w][3][lane][r] + (float)gx[gb + 3 * Hd + n];
      float cold = c[(size_t)m * Hd + n];
      float cn = fsigmoid(fv) * cold + fsigmoid(iv) * ftanh(gv);
      float hn = fsigmoid(ov) * ftanh(cn);
      c[(size_t)m * Hd + n] = cn;
      hout[(size_t)m * Hd + n] = (__bf16)hn;
      if (hs) hs[(size_t)m * Hd + n] = (__bf16)hn;
    }
  }
}

// ---------------- host side -------------------------------------------------
extern "C" void kernel_launch(void* const* d_in, const int* in_sizes, int n_in,
                              void* d_out, int out_size, void* d_ws, size_t ws_size,
                              hipStream_t stream) {
  const int*   src     = (const int*)d_in[0];
  const int*   trg     = (const int*)d_in[1];
  const float* E_in    = (const float*)d_in[2];
  const float* E_out   = (const float*)d_in[3];
  const float* enc_Wih = (const float*)d_in[4];
  const float* enc_Whh = (const float*)d_in[5];
  const float* enc_bih = (const float*)d_in[6];
  const float* enc_bhh = (const float*)d_in[7];
  const float* dec_Wih = (const float*)d_in[8];
  const float* dec_Whh = (const float*)d_in[9];
  const float* dec_bih = (const float*)d_in[10];
  const float* dec_bhh = (const float*)d_in[11];
  const float* fc_W    = (const float*)d_in[12];
  const float* fc_b    = (const float*)d_in[13];
  float* out = (float*)d_out;

  char* ws = (char*)d_ws;
  size_t off = 0;
  auto carve = [&](size_t bytes) -> char* {
    char* p = ws + off;
    off = (off + bytes + 255) & ~(size_t)255;
    return p;
  };
  __bf16* encWih_b = (__bf16*)carve((size_t)Gd * Hd * 2);
  __bf16* encWhh_b = (__bf16*)carve((size_t)Gd * Hd * 2);
  __bf16* decWih_b = (__bf16*)carve((size_t)Gd * Hd * 2);
  __bf16* decWhh_b = (__bf16*)carve((size_t)Gd * Hd * 2);
  __bf16* fcW_b    = (__bf16*)carve((size_t)Vd * Hd * 2);
  __bf16* src_emb  = (__bf16*)carve((size_t)Sd * Bd * Hd * 2);
  __bf16* dec_in   = (__bf16*)carve((size_t)(Td - 1) * Bd * Hd * 2);
  __bf16* gx_enc   = (__bf16*)carve((size_t)Sd * Bd * Gd * 2);
  __bf16* gx_dec   = (__bf16*)carve((size_t)(Td - 1) * Bd * Gd * 2);
  __bf16* hbuf0    = (__bf16*)carve((size_t)Bd * Hd * 2);
  __bf16* hbuf1    = (__bf16*)carve((size_t)Bd * Hd * 2);
  float*  cbuf     = (float*)carve((size_t)Bd * Hd * 4);
  __bf16* hs_dec   = (__bf16*)carve((size_t)(Td - 1) * Bd * Hd * 2);
  __bf16* hbuf[2]  = {hbuf0, hbuf1};

  // 1) weight conversions fp32 -> bf16
  k_f32_to_bf16<<<2048, 256, 0, stream>>>(enc_Wih, encWih_b, (long)Gd * Hd);
  k_f32_to_bf16<<<2048, 256, 0, stream>>>(enc_Whh, encWhh_b, (long)Gd * Hd);
  k_f32_to_bf16<<<2048, 256, 0, stream>>>(dec_Wih, decWih_b, (long)Gd * Hd);
  k_f32_to_bf16<<<2048, 256, 0, stream>>>(dec_Whh, decWhh_b, (long)Gd * Hd);
  k_f32_to_bf16<<<4096, 256, 0, stream>>>(fc_W, fcW_b, (long)Vd * Hd);

  // 2) embedding gathers
  k_embed<<<2048, 256, 0, stream>>>(src, Sd, E_in, src_emb, (long)Sd * Bd * Hd);
  k_embed<<<2048, 256, 0, stream>>>(trg, Td, E_out, dec_in, (long)(Td - 1) * Bd * Hd);

  // 3) init state, zero out[:,0,:]
  k_zero_bf16<<<64, 256, 0, stream>>>(hbuf0, (long)Bd * Hd);
  k_zero_f32 <<<64, 256, 0, stream>>>(cbuf, (long)Bd * Hd);
  k_zero_t0  <<<512, 256, 0, stream>>>(out);

  // 4) time-parallel input-gate GEMMs: gx = X @ Wih^T + (bih + bhh)
  {
    int M = Sd * Bd, Mt = (M + 31) / 32, Nt = Gd / 64;
    int blocks = (Mt * Nt + 7) / 8;
    k_gemm32<<<blocks, 256, 0, stream>>>(src_emb, encWih_b, enc_bih, enc_bhh,
                                         gx_enc, M, Mt, Nt);
  }
  {
    int M = (Td - 1) * Bd, Mt = (M + 31) / 32, Nt = Gd / 64;
    int blocks = (Mt * Nt + 7) / 8;
    k_gemm32<<<blocks, 256, 0, stream>>>(dec_in, decWih_b, dec_bih, dec_bhh,
                                         gx_dec, M, Mt, Nt);
  }

  // 5) encoder recurrence (512 steps), then decoder (255 steps)
  int flip = 0;
  for (int t = 0; t < Sd; ++t) {
    k_lstm_step<<<32, 256, 0, stream>>>(gx_enc + (size_t)t * Bd * Gd,
                                        hbuf[flip], encWhh_b, cbuf,
                                        hbuf[flip ^ 1], (__bf16*)nullptr);
    flip ^= 1;
  }
  for (int t = 0; t < Td - 1; ++t) {
    k_lstm_step<<<32, 256, 0, stream>>>(gx_dec + (size_t)t * Bd * Gd,
                                        hbuf[flip], decWhh_b, cbuf,
                                        hbuf[flip ^ 1],
                                        hs_dec + (size_t)t * Bd * Hd);
    flip ^= 1;
  }

  // 6) output projection: (4080 x 1024) x (1024 x 32000) + bias -> (B,T,V)
  {
    int M = (Td - 1) * Bd, Mt = (M + 31) / 32, Nt = Vd / 64;
    int blocks = (Mt * Nt + 7) / 8;
    k_fc32<<<blocks, 256, 0, stream>>>(hs_dec, fcW_b, fc_b, out, M, Mt, Nt);
  }
}
